// RoPEAttention_13357348291165
// MI455X (gfx1250) — matrix operand order repaired and checked by reading, not compiled, not run on hardware
//
#include <hip/hip_runtime.h>
#include <hip/hip_bf16.h>

// ---------------------------------------------------------------------------
// RoPE attention forward for MI455X (gfx1250), bf16 WMMA path + TDM tile DMA.
//   B=4, N=2048, C=1024, H=16, hd=64, AXIS_DIM=20, GRID=16
// Workspace layout (needs >= 72 MB):
//   [ 0MB) x_bf16      8192x1024        (reused as attn_out after QKV gemm)
//   [16MB) w_qkv_bf16  3072x1024
//   [22MB) w_proj_bf16 1024x1024
//   [24MB) q  [B,H,N,64] bf16
//   [40MB) k  [B,H,N,64] bf16
//   [56MB) vT [B,H,64,N] bf16
// ---------------------------------------------------------------------------

typedef __attribute__((ext_vector_type(16))) __bf16       bf16x16;
typedef __attribute__((ext_vector_type(8)))  float        f32x8;
typedef __attribute__((ext_vector_type(4)))  unsigned int u32x4;
typedef __attribute__((ext_vector_type(8)))  int          i32x8;
typedef __attribute__((ext_vector_type(4)))  int          i32x4;

#define DEV static __device__ __forceinline__

#if __has_builtin(__builtin_amdgcn_tensor_load_to_lds) && \
    __has_builtin(__builtin_amdgcn_s_wait_tensorcnt)
#define USE_TDM 1
#else
#define USE_TDM 0
#endif

DEV unsigned short f2bf(float f) {            // RNE float -> bf16 bits
  unsigned u = __builtin_bit_cast(unsigned, f);
  u += 0x7FFFu + ((u >> 16) & 1u);
  return (unsigned short)(u >> 16);
}
DEV float bf2f(unsigned short u) {
  unsigned x = ((unsigned)u) << 16;
  return __builtin_bit_cast(float, x);
}

union V16 { bf16x16 v; u32x4 q[2]; };

// A-operand (16xK bf16, K-chunk of 32): lane holds row (lane&15);
// halves [cb..cb+7] -> elems 0..7, [cb+16..cb+23] -> elems 8..15, cb=(lane>>4)*8.
DEV bf16x16 load_opA(const unsigned short* rowp, int kc, int lane) {
  int cb = kc + ((lane >> 4) << 3);
  V16 u;
  u.q[0] = *(const u32x4*)(rowp + cb);
  u.q[1] = *(const u32x4*)(rowp + cb + 16);
  return u.v;
}
// B-operand (Kx16 bf16 given as Bt[n][k]): lane holds col (lane&15);
// halves [cb..cb+7] -> elems 0..7, [cb+16..cb+23] -> elems 8..15, cb=(lane>>4)*8.
DEV bf16x16 load_opB(const unsigned short* rowp, int kc, int lane) {
  int cb = kc + ((lane >> 4) << 3);
  V16 u;
  u.q[0] = *(const u32x4*)(rowp + cb);
  u.q[1] = *(const u32x4*)(rowp + cb + 16);
  return u.v;
}

DEV f32x8 wmma_bf16(bf16x16 a, bf16x16 b, f32x8 c) {
  return __builtin_amdgcn_wmma_f32_16x16x32_bf16(
      /*neg_a=*/false, a, /*neg_b=*/false, b,
      /*c_mod=*/(short)0, c, /*reuse_a=*/false, /*reuse_b=*/false);
}

// LDS byte offset of a __shared__ object: generic-pointer low 32 bits
DEV unsigned lds_off(const void* p) { return (unsigned)(unsigned long long)p; }

// ---------------------------------------------------------------------------
// TDM: 2D tile DMA global -> LDS, data_size = 4B elements, LDS row padding.
//   tileWdw : tile width in dwords        tileH : tile height (rows)
//   strideDw: tensor row stride in dwords
//   padIntC : pad_interval code (0=2,1=4,2=8,3=16,4=32,... dwords)
//   padAmtC : pad_amount code   (dwords-1)
// D# bitfields per CDNA5 ISA 8.3/8.4 (count=1, type=2, pad_enable=1).
// ---------------------------------------------------------------------------
DEV void tdm_load_2d(unsigned ldsAddr, const void* gptr,
                     unsigned tileWdw, unsigned tileH, unsigned strideDw,
                     unsigned padIntC, unsigned padAmtC) {
#if USE_TDM
  unsigned long long ga = (unsigned long long)gptr;
  u32x4 g0;
  g0.x = 1u;                                        // count=1, user mode
  g0.y = ldsAddr;                                   // lds_addr (bytes)
  g0.z = (unsigned)ga;                              // global_addr[31:0]
  g0.w = (unsigned)((ga >> 32) & 0x01FFFFFFu)       // global_addr[56:32]
         | 0x80000000u;                             // type=2 ("image")
  i32x8 g1;
  g1[0] = (int)((2u << 16)                          // data_size = 4B
                | (1u << 20)                        // pad_enable
                | (padIntC << 22) | (padAmtC << 25));
  g1[1] = (int)((tileWdw & 0xFFFFu) << 16);         // tensor_dim0[15:0]
  g1[2] = (int)(((tileWdw >> 16) & 0xFFFFu)         // tensor_dim0[31:16]
                | ((tileH & 0xFFFFu) << 16));       // tensor_dim1[15:0]
  g1[3] = (int)(((tileH >> 16) & 0xFFFFu)           // tensor_dim1[31:16]
                | ((tileWdw & 0xFFFFu) << 16));     // tile_dim0
  g1[4] = (int)(tileH & 0xFFFFu);                   // tile_dim1 (tile_dim2=0)
  g1[5] = (int)strideDw;                            // tensor_dim0_stride[31:0]
  g1[6] = 0;
  g1[7] = 0;
  i32x4 z4 = {0, 0, 0, 0};
#if __clang_major__ >= 23
  i32x8 z8 = {0, 0, 0, 0, 0, 0, 0, 0};
  __builtin_amdgcn_tensor_load_to_lds(g0, g1, z4, z4, z8, 0);
#else
  __builtin_amdgcn_tensor_load_to_lds(g0, g1, z4, z4, 0);
#endif
#else
  (void)ldsAddr; (void)gptr; (void)tileWdw; (void)tileH; (void)strideDw;
  (void)padIntC; (void)padAmtC;
#endif
}

// ---------------------------------------------------------------------------
// fp32 -> bf16 convert
// ---------------------------------------------------------------------------
__global__ __launch_bounds__(256) void k_cvt(const float* __restrict__ in,
                                             unsigned short* __restrict__ out,
                                             int n) {
  int i = blockIdx.x * 256 + threadIdx.x;
  if (i < n) out[i] = f2bf(in[i]);
}

// ---------------------------------------------------------------------------
// 128x128 block GEMM: C = A[M,K=1024] * Bt[N,K=1024]^T
// 256 threads = 8 waves; wave tile 32x64 = 2x4 WMMA accumulators.
// ---------------------------------------------------------------------------
#define LDT 40       // padded LDS row stride (halves) = 16+4 dwords
#define GTILE (128 * LDT)

DEV void gemm_compute(const unsigned short* Ac, const unsigned short* Bc,
                      f32x8 acc[2][4], int lane, int wm, int wn) {
  bf16x16 av[2], bv[4];
#pragma unroll
  for (int mt = 0; mt < 2; ++mt)
    av[mt] = load_opA(Ac + (wm + mt * 16 + (lane & 15)) * LDT, 0, lane);
#pragma unroll
  for (int nt = 0; nt < 4; ++nt)
    bv[nt] = load_opB(Bc + (wn + nt * 16 + (lane & 15)) * LDT, 0, lane);
#pragma unroll
  for (int mt = 0; mt < 2; ++mt)
#pragma unroll
    for (int nt = 0; nt < 4; ++nt)
      acc[mt][nt] = wmma_bf16(av[mt], bv[nt], acc[mt][nt]);
}

DEV void gemm_mainloop(const unsigned short* __restrict__ A,
                       const unsigned short* __restrict__ Bt,
                       int bm, int bn,
                       unsigned short* As, unsigned short* Bs,
                       f32x8 acc[2][4]) {
  const int K = 1024;
  const int tid = threadIdx.x, lane = tid & 31, wid = tid >> 5;
  const int wm = (wid & 3) * 32, wn = (wid >> 2) * 64;
  const f32x8 zero = {0.f, 0.f, 0.f, 0.f, 0.f, 0.f, 0.f, 0.f};
#pragma unroll
  for (int mt = 0; mt < 2; ++mt)
#pragma unroll
    for (int nt = 0; nt < 4; ++nt) acc[mt][nt] = zero;

#if USE_TDM
  // double-buffered TDM tile DMA, issued by wave 0, tracked by TENSORcnt
  const int KD = K >> 1;  // row stride in dwords
  if (wid == 0) {
    tdm_load_2d(lds_off(As), A + (size_t)bm * K, 16, 128, KD, 3, 3);
    tdm_load_2d(lds_off(Bs), Bt + (size_t)bn * K, 16, 128, KD, 3, 3);
  }
  int cur = 0;
  for (int it = 0; it < 32; ++it) {
    if (wid == 0) {
      if (it < 31) {
        int nxt = cur ^ 1;
        tdm_load_2d(lds_off(As + nxt * GTILE),
                    A + (size_t)bm * K + (it + 1) * 32, 16, 128, KD, 3, 3);
        tdm_load_2d(lds_off(Bs + nxt * GTILE),
                    Bt + (size_t)bn * K + (it + 1) * 32, 16, 128, KD, 3, 3);
        __builtin_amdgcn_s_wait_tensorcnt(2);   // current buffer complete
      } else {
        __builtin_amdgcn_s_wait_tensorcnt(0);
      }
    }
    __syncthreads();
    gemm_compute(As + cur * GTILE, Bs + cur * GTILE, acc, lane, wm, wn);
    __syncthreads();
    cur ^= 1;
  }
#else
  // fallback: cooperative VGPR staging (single buffer)
  const int lrow = tid >> 1, lk = (tid & 1) * 16;
  for (int kk = 0; kk < K; kk += 32) {
    __syncthreads();
    const u32x4* ga = (const u32x4*)(A + (size_t)(bm + lrow) * K + kk + lk);
    u32x4 a0 = ga[0], a1 = ga[1];
    const u32x4* gb = (const u32x4*)(Bt + (size_t)(bn + lrow) * K + kk + lk);
    u32x4 b0 = gb[0], b1 = gb[1];
    *(u32x4*)(As + lrow * LDT + lk)     = a0;
    *(u32x4*)(As + lrow * LDT + lk + 8) = a1;
    *(u32x4*)(Bs + lrow * LDT + lk)     = b0;
    *(u32x4*)(Bs + lrow * LDT + lk + 8) = b1;
    __syncthreads();
    gemm_compute(As, Bs, acc, lane, wm, wn);
  }
#endif
}

// ---------------------------------------------------------------------------
// QKV GEMM: epilogue scatters bf16 into q[B,H,N,64], k[B,H,N,64], vT[B,H,64,N]
// ---------------------------------------------------------------------------
__global__ __launch_bounds__(256) void k_gemm_qkv(
    const unsigned short* __restrict__ A, const unsigned short* __restrict__ Bt,
    unsigned short* __restrict__ qb, unsigned short* __restrict__ kb,
    unsigned short* __restrict__ vT) {
  __shared__ unsigned short As[2 * GTILE];
  __shared__ unsigned short Bs[2 * GTILE];
  const int bm = blockIdx.y * 128, bn = blockIdx.x * 128;
  f32x8 acc[2][4];
  gemm_mainloop(A, Bt, bm, bn, As, Bs, acc);

  const int lane = threadIdx.x & 31, wid = threadIdx.x >> 5;
  const int wm = (wid & 3) * 32, wn = (wid >> 2) * 64;
#pragma unroll
  for (int mt = 0; mt < 2; ++mt)
#pragma unroll
    for (int nt = 0; nt < 4; ++nt)
#pragma unroll
      for (int r = 0; r < 8; ++r) {
        int m = bm + wm + mt * 16 + r + ((lane >> 4) << 3);
        int j = bn + wn + nt * 16 + (lane & 15);
        unsigned short val = f2bf(acc[mt][nt][r]);
        int b = m >> 11, n = m & 2047;
        int part = j >> 10, c = j & 1023, head = c >> 6, ch = c & 63;
        int bh = b * 16 + head;
        if (part == 0)      qb[((size_t)bh * 2048 + n) * 64 + ch] = val;
        else if (part == 1) kb[((size_t)bh * 2048 + n) * 64 + ch] = val;
        else                vT[((size_t)bh * 64 + ch) * 2048 + n] = val;
      }
}

// ---------------------------------------------------------------------------
// Projection GEMM: + bias -> fp32 out
// ---------------------------------------------------------------------------
__global__ __launch_bounds__(256) void k_gemm_proj(
    const unsigned short* __restrict__ A, const unsigned short* __restrict__ Bt,
    const float* __restrict__ bias, float* __restrict__ out) {
  __shared__ unsigned short As[2 * GTILE];
  __shared__ unsigned short Bs[2 * GTILE];
  const int bm = blockIdx.y * 128, bn = blockIdx.x * 128;
  f32x8 acc[2][4];
  gemm_mainloop(A, Bt, bm, bn, As, Bs, acc);

  const int lane = threadIdx.x & 31, wid = threadIdx.x >> 5;
  const int wm = (wid & 3) * 32, wn = (wid >> 2) * 64;
#pragma unroll
  for (int mt = 0; mt < 2; ++mt)
#pragma unroll
    for (int nt = 0; nt < 4; ++nt)
#pragma unroll
      for (int r = 0; r < 8; ++r) {
        int m = bm + wm + mt * 16 + r + ((lane >> 4) << 3);
        int j = bn + wn + nt * 16 + (lane & 15);
        out[(size_t)m * 1024 + j] = acc[mt][nt][r] + bias[j];
      }
}

// ---------------------------------------------------------------------------
// RoPE in place on q/k (bf16), channels [0,60): 3 axes x 10 freqs.
// ---------------------------------------------------------------------------
__global__ __launch_bounds__(256) void k_rope(unsigned short* __restrict__ qb,
                                              unsigned short* __restrict__ kb) {
  int idx = blockIdx.x * 256 + threadIdx.x;
  if (idx >= 64 * 2048 * 30) return;
  int pair = idx % 30;
  int row  = idx / 30;              // row = bh*2048 + n
  int n    = row & 2047;
  int axis = pair / 10, i = pair % 10;
  float pos = (axis == 0) ? (float)(n >> 8)
            : (axis == 1) ? (float)((n >> 4) & 15)
                          : (float)(n & 15);
  float omega = __powf(10000.0f, -(float)i * 0.1f);
  float ang = pos * omega;
  float s = __sinf(ang), c = __cosf(ang);
  size_t off = (size_t)row * 64 + axis * 20 + i * 2;
  float x1 = bf2f(qb[off]), x2 = bf2f(qb[off + 1]);
  qb[off]     = f2bf(x1 * c - x2 * s);
  qb[off + 1] = f2bf(x2 * c + x1 * s);
  x1 = bf2f(kb[off]); x2 = bf2f(kb[off + 1]);
  kb[off]     = f2bf(x1 * c - x2 * s);
  kb[off + 1] = f2bf(x2 * c + x1 * s);
}

// ---------------------------------------------------------------------------
// Flash attention: grid = B*H*16 blocks; 8 waves/block, 16 q-rows per wave.
// Key blocks of 64; TDM double-buffered K/V tiles; per-wave P staging for
// the C->A layout transpose; online softmax with 16-lane shuffles.
// ---------------------------------------------------------------------------
#define LDK 72       // padded row stride (halves) = 32+4 dwords
#define KTILE (64 * LDK)

DEV void attn_block(const unsigned short* Ksb, const unsigned short* Vsb,
                    unsigned short* pw, const bf16x16 aq[2],
                    f32x8 acc[4], float mrow[8], float lrow[8], int lane) {
  const f32x8 zero = {0.f, 0.f, 0.f, 0.f, 0.f, 0.f, 0.f, 0.f};
  // scores S = (Q * K^T) * scale : 4 col-tiles x 2 K-chunks
  f32x8 s[4];
#pragma unroll
  for (int nt = 0; nt < 4; ++nt) {
    s[nt] = zero;
#pragma unroll
    for (int kc = 0; kc < 2; ++kc) {
      bf16x16 bk = load_opB(Ksb + (nt * 16 + (lane & 15)) * LDK, kc * 32, lane);
      s[nt] = wmma_bf16(aq[kc], bk, s[nt]);
    }
  }
  // online softmax update per row
#pragma unroll
  for (int r = 0; r < 8; ++r) {
    float s0 = s[0][r] * 0.125f, s1 = s[1][r] * 0.125f;
    float s2 = s[2][r] * 0.125f, s3 = s[3][r] * 0.125f;
    float mx = fmaxf(fmaxf(s0, s1), fmaxf(s2, s3));
    mx = fmaxf(mx, __shfl_xor(mx, 1, 16));
    mx = fmaxf(mx, __shfl_xor(mx, 2, 16));
    mx = fmaxf(mx, __shfl_xor(mx, 4, 16));
    mx = fmaxf(mx, __shfl_xor(mx, 8, 16));
    float mnew = fmaxf(mrow[r], mx);
    float p0 = __expf(s0 - mnew), p1 = __expf(s1 - mnew);
    float p2 = __expf(s2 - mnew), p3 = __expf(s3 - mnew);
    float rs = p0 + p1 + p2 + p3;
    rs += __shfl_xor(rs, 1, 16);
    rs += __shfl_xor(rs, 2, 16);
    rs += __shfl_xor(rs, 4, 16);
    rs += __shfl_xor(rs, 8, 16);
    float alpha = __expf(mrow[r] - mnew);
    mrow[r] = mnew;
    lrow[r] = lrow[r] * alpha + rs;
#pragma unroll
    for (int ct = 0; ct < 4; ++ct) acc[ct][r] *= alpha;
    s[0][r] = p0; s[1][r] = p1; s[2][r] = p2; s[3][r] = p3;
  }
  // stage P (C-layout -> LDS bf16), reload in A-layout (same-wave LDS in order)
#pragma unroll
  for (int nt = 0; nt < 4; ++nt)
#pragma unroll
    for (int r = 0; r < 8; ++r)
      pw[(r + ((lane >> 4) << 3)) * LDK + nt * 16 + (lane & 15)] = f2bf(s[nt][r]);
#pragma unroll
  for (int kc = 0; kc < 2; ++kc) {
    bf16x16 pa = load_opA(pw + (lane & 15) * LDK, kc * 32, lane);
#pragma unroll
    for (int ct = 0; ct < 4; ++ct) {
      bf16x16 bv = load_opB(Vsb + (ct * 16 + (lane & 15)) * LDK, kc * 32, lane);
      acc[ct] = wmma_bf16(pa, bv, acc[ct]);
    }
  }
}

__global__ __launch_bounds__(256) void k_attn(
    const unsigned short* __restrict__ qg, const unsigned short* __restrict__ kg,
    const unsigned short* __restrict__ vg, unsigned short* __restrict__ outb) {
  __shared__ unsigned short Ks[2 * KTILE];
  __shared__ unsigned short Vs[2 * KTILE];
  __shared__ unsigned short Ps[8 * 16 * LDK];

  const int tid = threadIdx.x, lane = tid & 31, wid = tid >> 5;
  const int bh = blockIdx.x >> 4, qblk = blockIdx.x & 15;
  const int b = bh >> 4, h = bh & 15;
  const unsigned short* qp = qg + (size_t)bh * 2048 * 64;
  const unsigned short* kp = kg + (size_t)bh * 2048 * 64;
  const unsigned short* vp = vg + (size_t)bh * 64 * 2048;
  const int q0 = qblk * 128 + wid * 16;

  bf16x16 aq[2];
  {
    const unsigned short* rowp = qp + (size_t)(q0 + (lane & 15)) * 64;
    aq[0] = load_opA(rowp, 0, lane);
    aq[1] = load_opA(rowp, 32, lane);
  }
  float mrow[8], lrow[8];
  f32x8 acc[4];
  const f32x8 zero = {0.f, 0.f, 0.f, 0.f, 0.f, 0.f, 0.f, 0.f};
#pragma unroll
  for (int r = 0; r < 8; ++r) { mrow[r] = -1e30f; lrow[r] = 0.f; }
#pragma unroll
  for (int ct = 0; ct < 4; ++ct) acc[ct] = zero;

  unsigned short* pw = Ps + wid * (16 * LDK);

#if USE_TDM
  if (wid == 0) {
    tdm_load_2d(lds_off(Ks), kp, 32, 64, 32, 4, 3);
    tdm_load_2d(lds_off(Vs), vp, 32, 64, 1024, 4, 3);
  }
  int buf = 0;
  for (int kbv = 0; kbv < 32; ++kbv) {
    if (wid == 0) {
      if (kbv < 31) {
        int nxt = buf ^ 1;
        tdm_load_2d(lds_off(Ks + nxt * KTILE),
                    kp + (size_t)(kbv + 1) * 64 * 64, 32, 64, 32, 4, 3);
        tdm_load_2d(lds_off(Vs + nxt * KTILE),
                    vp + (size_t)(kbv + 1) * 64, 32, 64, 1024, 4, 3);
        __builtin_amdgcn_s_wait_tensorcnt(2);
      } else {
        __builtin_amdgcn_s_wait_tensorcnt(0);
      }
    }
    __syncthreads();
    attn_block(Ks + buf * KTILE, Vs + buf * KTILE, pw, aq, acc, mrow, lrow, lane);
    __syncthreads();
    buf ^= 1;
  }
#else
  const int lr = tid >> 2, lseg = (tid & 3) * 16;
  for (int kbv = 0; kbv < 32; ++kbv) {
    __syncthreads();
    const u32x4* gk = (const u32x4*)(kp + (size_t)(kbv * 64 + lr) * 64 + lseg);
    u32x4 k0 = gk[0], k1 = gk[1];
    const u32x4* gv = (const u32x4*)(vp + (size_t)lr * 2048 + kbv * 64 + lseg);
    u32x4 v0 = gv[0], v1 = gv[1];
    *(u32x4*)(Ks + lr * LDK + lseg)     = k0;
    *(u32x4*)(Ks + lr * LDK + lseg + 8) = k1;
    *(u32x4*)(Vs + lr * LDK + lseg)     = v0;
    *(u32x4*)(Vs + lr * LDK + lseg + 8) = v1;
    __syncthreads();
    attn_block(Ks, Vs, pw, aq, acc, mrow, lrow, lane);
  }
#endif

  // normalize and write attn_out[b,n,h*64+ch] as bf16 (A of proj GEMM)
#pragma unroll
  for (int ct = 0; ct < 4; ++ct)
#pragma unroll
    for (int r = 0; r < 8; ++r) {
      float o = acc[ct][r] / lrow[r];
      int m = r + ((lane >> 4) << 3);
      int ch = ct * 16 + (lane & 15);
      outb[((size_t)b * 2048 + q0 + m) * 1024 + h * 64 + ch] = f2bf(o);
    }
}

// ---------------------------------------------------------------------------
extern "C" void kernel_launch(void* const* d_in, const int* in_sizes, int n_in,
                              void* d_out, int out_size, void* d_ws, size_t ws_size,
                              hipStream_t stream) {
  (void)in_sizes; (void)n_in; (void)out_size; (void)ws_size;
  const float* x      = (const float*)d_in[0];   // [4,2048,1024]
  const float* w_qkv  = (const float*)d_in[1];   // [3072,1024]
  const float* w_proj = (const float*)d_in[2];   // [1024,1024]
  const float* b_proj = (const float*)d_in[3];   // [1024]
  float* out = (float*)d_out;                    // [4,2048,1024] fp32

  char* ws = (char*)d_ws;
  unsigned short* xb  = (unsigned short*)(ws);
  unsigned short* wqb = (unsigned short*)(ws + (16u << 20));
  unsigned short* wpb = (unsigned short*)(ws + (22u << 20));
  unsigned short* qb  = (unsigned short*)(ws + (24u << 20));
  unsigned short* kb  = (unsigned short*)(ws + (40u << 20));
  unsigned short* vT  = (unsigned short*)(ws + (56u << 20));
  unsigned short* attn = xb;  // xb dead after QKV GEMM; stream order makes this safe

  k_cvt<<<8388608 / 256, 256, 0, stream>>>(x, xb, 8388608);
  k_cvt<<<3145728 / 256, 256, 0, stream>>>(w_qkv, wqb, 3145728);
  k_cvt<<<1048576 / 256, 256, 0, stream>>>(w_proj, wpb, 1048576);

  k_gemm_qkv<<<dim3(24, 64), 256, 0, stream>>>(xb, wqb, qb, kb, vT);
  k_rope<<<15360, 256, 0, stream>>>(qb, kb);
  k_attn<<<1024, 256, 0, stream>>>(qb, kb, vT, attn);
  k_gemm_proj<<<dim3(8, 64), 256, 0, stream>>>(attn, wpb, b_proj, out);
}
